// AdaptiveSpanAttention_66846870995239
// MI455X (gfx1250) — matrix operand order repaired and checked by reading, not compiled, not run on hardware
//
#include <hip/hip_runtime.h>
#include <hip/hip_bf16.h>

typedef __attribute__((ext_vector_type(16))) _Float16 v16h;
typedef __attribute__((ext_vector_type(8)))  _Float16 v8h_t;
typedef __attribute__((ext_vector_type(8)))  float    v8f;
typedef __attribute__((ext_vector_type(4)))  int      v4i;

#define D_MODEL 1024
#define NHEAD   16
#define DHEAD   64
#define TSEQ    2048
#define BATCH   2
#define MTOT    (BATCH*TSEQ)   // 4096

// ---------------------------------------------------------------------------
// Optional CDNA5 async global->LDS copy (ASYNCcnt) with sync fallback.
// Builtin signature (from hipcc diagnostic): (v4i AS1*, v4i AS3*, imm, imm)
// ---------------------------------------------------------------------------
#ifndef __has_builtin
#define __has_builtin(x) 0
#endif
#if __has_builtin(__builtin_amdgcn_global_load_async_to_lds_b128) && \
    __has_builtin(__builtin_amdgcn_s_wait_asynccnt)
#define USE_ASYNC_LDS 1
#else
#define USE_ASYNC_LDS 0
#endif

__device__ inline void lds_copy16(const _Float16* g, _Float16* l) {
#if USE_ASYNC_LDS
  void* gv = const_cast<_Float16*>(g);
  __builtin_amdgcn_global_load_async_to_lds_b128(
      (__attribute__((address_space(1))) v4i*)gv,
      (__attribute__((address_space(3))) v4i*)l, 0, 0);
#else
  *reinterpret_cast<v8h_t*>(l) = *reinterpret_cast<const v8h_t*>(g);
#endif
}
__device__ inline void async_fence() {
#if USE_ASYNC_LDS
  __builtin_amdgcn_s_wait_asynccnt(0);
#endif
}

// ---------------------------------------------------------------------------
// Fragment loaders (wave32 WMMA 16x16x32 f16; contiguous k order, same for A and B)
// A-frag: lanes0-15 row M=l15 hold K = 0..15; lanes16-31 hold K = 16..31 (kb = half*8)
// B-frag: lanes0-15 col N=l15 hold K = 0..15; lanes16-31 hold K = 16..31
// ---------------------------------------------------------------------------
__device__ inline v16h frag_a16(const _Float16* rowp, int kbase) {
  v8h_t x0 = *reinterpret_cast<const v8h_t*>(rowp + 2 * kbase);
  v8h_t x1 = *reinterpret_cast<const v8h_t*>(rowp + 2 * kbase + 8);
  v16h r;
#pragma unroll
  for (int i = 0; i < 8; ++i) { r[i] = x0[i]; r[8 + i] = x1[i]; }
  return r;
}
__device__ inline v16h frag_b16(const _Float16* p) {  // 16 contiguous halves
  v8h_t x0 = *reinterpret_cast<const v8h_t*>(p);
  v8h_t x1 = *reinterpret_cast<const v8h_t*>(p + 8);
  v16h r;
#pragma unroll
  for (int i = 0; i < 8; ++i) { r[i] = x0[i]; r[8 + i] = x1[i]; }
  return r;
}

// ---------------------------------------------------------------------------
// small prep kernels
// ---------------------------------------------------------------------------
__global__ void f32_to_f16(const float* __restrict__ in, _Float16* __restrict__ out, int n) {
  int i = blockIdx.x * 256 + threadIdx.x;
  if (i < n) out[i] = (_Float16)in[i];
}

__global__ void colmean_kernel(const float* __restrict__ x, float* __restrict__ xmean) {
  int i = blockIdx.x * 256 + threadIdx.x;
  if (i >= BATCH * D_MODEL) return;
  int b = i >> 10, d = i & (D_MODEL - 1);
  const float* p = x + (size_t)b * TSEQ * D_MODEL + d;
  float s = 0.f;
  for (int t = 0; t < TSEQ; ++t) s += p[(size_t)t * D_MODEL];
  xmean[i] = s * (1.0f / (float)TSEQ);
}

__global__ void span_z_kernel(const float* __restrict__ xmean, const float* __restrict__ span_w,
                              const float* __restrict__ span_b, float* __restrict__ z) {
  int i = threadIdx.x;
  if (i >= BATCH * NHEAD) return;
  int b = i / NHEAD, h = i % NHEAD;
  float acc = span_b[h];
  for (int d = 0; d < D_MODEL; ++d) acc += xmean[b * D_MODEL + d] * span_w[d * NHEAD + h];
  z[i] = (float)TSEQ / (1.0f + __expf(-acc));
}

// ---------------------------------------------------------------------------
// WMMA GEMM: C(4096x1024) = A(f16) * B(f16), K=1024.
// Block tile 128x128, 8 waves, wave tile 64x32, K-step 32.
// Double-buffered LDS (stage tile k+32 while computing tile k, 1 barrier/iter);
// the A tile moves via global_load_async_to_lds_b128 (ASYNCcnt-tracked DMA).
// B tile transposed in LDS so B-fragments read contiguously.
// MODE 1: f16 (B,H,T,DH); MODE 2: f16 (B,H,DH,T); MODE 3: f32 + bias; else f16 RM.
// ---------------------------------------------------------------------------
template<int MODE>
__global__ __launch_bounds__(256)
void gemm128_wmma(const _Float16* __restrict__ A, const _Float16* __restrict__ Bm,
                  const float* __restrict__ bias, void* __restrict__ out)
{
  __shared__ _Float16 As[2][128 * 40];   // [m][k], stride 40 halves (80B)
  __shared__ _Float16 Bt[2][128 * 40];   // [n][k] transposed

  const int tid  = threadIdx.x;
  const int wave = tid >> 5, lane = tid & 31;
  const int hh   = lane >> 4, l15 = lane & 15;
  const int m0   = blockIdx.x * 128, n0 = blockIdx.y * 128;
  const int wm   = (wave >> 2) * 64, wn = (wave & 3) * 32;

  const int rowA = tid >> 2, kcA = tid & 3;   // + i*64  -> 128 rows x 4 chunks
  const int rowB = tid >> 4, ncB = tid & 15;  // + i*16  -> 32 rows  x 16 chunks

  auto stage = [&](int k0, int buf) {
#pragma unroll
    for (int i = 0; i < 2; ++i) {
      int rA = rowA + i * 64;
      lds_copy16(A + (size_t)(m0 + rA) * D_MODEL + k0 + kcA * 8,
                 &As[buf][rA * 40 + kcA * 8]);
    }
#pragma unroll
    for (int i = 0; i < 2; ++i) {
      int rB = rowB + i * 16;
      v8h_t v = *reinterpret_cast<const v8h_t*>(Bm + (size_t)(k0 + rB) * D_MODEL + n0 + ncB * 8);
#pragma unroll
      for (int j = 0; j < 8; ++j) Bt[buf][(ncB * 8 + j) * 40 + rB] = v[j];
    }
  };

  v8f acc[4][2];
#pragma unroll
  for (int t = 0; t < 4; ++t)
#pragma unroll
    for (int u = 0; u < 2; ++u)
#pragma unroll
      for (int r = 0; r < 8; ++r) acc[t][u][r] = 0.f;

  stage(0, 0);
  async_fence();
  __syncthreads();

  for (int k0 = 0; k0 < D_MODEL; k0 += 32) {
    const int cur = (k0 >> 5) & 1;
    if (k0 + 32 < D_MODEL) stage(k0 + 32, cur ^ 1);

    v16h af[4], bf[2];
#pragma unroll
    for (int t = 0; t < 4; ++t) af[t] = frag_a16(&As[cur][(wm + t * 16 + l15) * 40], hh * 8);
#pragma unroll
    for (int u = 0; u < 2; ++u) bf[u] = frag_b16(&Bt[cur][(wn + u * 16 + l15) * 40 + hh * 16]);
#pragma unroll
    for (int t = 0; t < 4; ++t)
#pragma unroll
      for (int u = 0; u < 2; ++u)
        acc[t][u] = __builtin_amdgcn_wmma_f32_16x16x32_f16(
            false, af[t], false, bf[u], (short)0, acc[t][u], false, false);

    async_fence();
    __syncthreads();
  }

  // Epilogue. C layout: VGPR r -> M = hh?8+r:r, N = l15.
#pragma unroll
  for (int t = 0; t < 4; ++t)
#pragma unroll
    for (int u = 0; u < 2; ++u) {
      int mb = m0 + wm + t * 16 + hh * 8;
      int n  = n0 + wn + u * 16 + l15;
#pragma unroll
      for (int r = 0; r < 8; ++r) {
        int m = mb + r;
        float val = acc[t][u][r];
        if constexpr (MODE == 1) {        // (B,H,T,DH) f16
          int b = m >> 11, tt = m & (TSEQ - 1), h = n >> 6, dh = n & (DHEAD - 1);
          ((_Float16*)out)[(((size_t)b * NHEAD + h) * TSEQ + tt) * DHEAD + dh] = (_Float16)val;
        } else if constexpr (MODE == 2) { // (B,H,DH,T) f16 (V transposed)
          int b = m >> 11, tt = m & (TSEQ - 1), h = n >> 6, dh = n & (DHEAD - 1);
          ((_Float16*)out)[(((size_t)b * NHEAD + h) * DHEAD + dh) * TSEQ + tt] = (_Float16)val;
        } else if constexpr (MODE == 3) { // f32 row-major + bias
          ((float*)out)[(size_t)m * D_MODEL + n] = val + bias[n];
        } else {                          // f16 row-major
          ((_Float16*)out)[(size_t)m * D_MODEL + n] = (_Float16)val;
        }
      }
    }
}

// ---------------------------------------------------------------------------
// Flash-style adaptive-span attention, S^T formulation.
//   S^T[key][q] = K[key]·Q[q]  -> each lane owns one query column, so the
//   online-softmax state is a per-lane scalar (one xor-16 shuffle to combine).
//   O^T[dh][q] += V^T[dh][key] * P[key][q]
// Block = 8 waves = 128 queries of one (b,h); K/V^T chunks (32 keys) are
// cooperatively staged in LDS (async DMA) once per block, shared by all waves.
// ---------------------------------------------------------------------------
__global__ __launch_bounds__(256)
void attn_span_wmma(const _Float16* __restrict__ Qg, const _Float16* __restrict__ Kg,
                    const _Float16* __restrict__ Vtg, const float* __restrict__ zg,
                    _Float16* __restrict__ Og)
{
  __shared__ _Float16 Ks[32 * 72];    // [key][dh]  stride 72 halves (144B)
  __shared__ _Float16 Vts[64 * 40];   // [dh][key]  stride 40 halves (80B)

  const int bh   = blockIdx.y;
  const int tid  = threadIdx.x;
  const int wave = tid >> 5, lane = tid & 31;
  const int hh   = lane >> 4, l15 = lane & 15;
  const int qb   = blockIdx.x * 128;
  const int q0   = qb + wave * 16;

  const _Float16* Q  = Qg  + (size_t)bh * TSEQ * DHEAD;
  const _Float16* K  = Kg  + (size_t)bh * TSEQ * DHEAD;
  const _Float16* Vt = Vtg + (size_t)bh * DHEAD * TSEQ;
  const float zz = zg[bh];
  const int qcol = q0 + l15;                 // this lane's query index

  const int krow = tid >> 3, kkc = tid & 7;  // K chunk: 32 rows x 8 chunks
  const int vrow = tid >> 2, vkc = tid & 3;  // Vt chunk: 64 rows x 4 chunks

  // Q^T B-fragments (dh chunks 0-31, 32-63): contiguous per lane, loop-invariant
  v16h qf[2];
#pragma unroll
  for (int d = 0; d < 2; ++d)
    qf[d] = frag_b16(Q + (size_t)qcol * DHEAD + d * 32 + hh * 16);

  v8f accO[4];
#pragma unroll
  for (int d = 0; d < 4; ++d)
#pragma unroll
    for (int r = 0; r < 8; ++r) accO[d][r] = 0.f;
  float mrun = -1e30f, lrun = 0.f;

  const int send_w   = (q0 + 16 < TSEQ) ? (q0 + 16) : TSEQ;   // wave's causal bound
  const int send_blk = (qb + 128 < TSEQ) ? (qb + 128) : TSEQ; // block's bound

  for (int s0 = 0; s0 < send_blk; s0 += 32) {
    // ---- cooperative stage of K[s0..s0+31][:] and Vt[:][s0..s0+31] ----
    lds_copy16(K  + (size_t)(s0 + krow) * DHEAD + kkc * 8, &Ks[krow * 72 + kkc * 8]);
    lds_copy16(Vt + (size_t)vrow * TSEQ + s0 + vkc * 8,    &Vts[vrow * 40 + vkc * 8]);
    if (s0 + 32 < send_blk) {   // prefetch next chunk into L2 (global_prefetch_b8)
      __builtin_prefetch((const void*)(K  + (size_t)(s0 + 32 + krow) * DHEAD + kkc * 8), 0, 1);
      __builtin_prefetch((const void*)(Vt + (size_t)vrow * TSEQ + s0 + 32 + vkc * 8), 0, 1);
    }
    async_fence();
    __syncthreads();

    if (s0 < send_w) {          // wave-uniform predicate (EXEC stays all-1 for WMMA)
      // ---- S^T = K * Q^T : two 16-key tiles, 2 wmma each (DH=64) ----
      v8f st[2];
#pragma unroll
      for (int t = 0; t < 2; ++t) {
#pragma unroll
        for (int r = 0; r < 8; ++r) st[t][r] = 0.f;
        const _Float16* krp = &Ks[(t * 16 + l15) * 72];
#pragma unroll
        for (int d = 0; d < 2; ++d) {
          v16h a = frag_a16(krp + d * 32, hh * 8);
          st[t] = __builtin_amdgcn_wmma_f32_16x16x32_f16(
              false, a, false, qf[d], (short)0, st[t], false, false);
        }
      }

      // ---- causal mask + online softmax (per-lane column) ----
      float sv[16];
      float cmax = -1e30f;
#pragma unroll
      for (int t = 0; t < 2; ++t)
#pragma unroll
        for (int r = 0; r < 8; ++r) {
          int key = s0 + t * 16 + hh * 8 + r;
          float s = st[t][r] * 0.125f;          // 1/sqrt(64)
          if (key > qcol) s = -1e30f;
          sv[t * 8 + r] = s;
          cmax = fmaxf(cmax, s);
        }
      cmax = fmaxf(cmax, __shfl_xor(cmax, 16));
      float mnew = fmaxf(mrun, cmax);
      float corr = __expf(mrun - mnew);

      float psum = 0.f;
      _Float16 ph[16];
#pragma unroll
      for (int t = 0; t < 2; ++t)
#pragma unroll
        for (int r = 0; r < 8; ++r) {
          int key = s0 + t * 16 + hh * 8 + r;
          float chi = (256.f + zz - (float)(qcol - key)) * (1.f / 256.f);
          chi = fminf(fmaxf(chi, 0.f), 1.f);    // soft span mask
          float p = (key <= qcol) ? __expf(sv[t * 8 + r] - mnew) * chi : 0.f;
          psum += p;
          ph[t * 8 + r] = (_Float16)p;
        }
      psum += __shfl_xor(psum, 16);
      lrun = lrun * corr + psum;
      mrun = mnew;
#pragma unroll
      for (int d = 0; d < 4; ++d)
#pragma unroll
        for (int r = 0; r < 8; ++r) accO[d][r] *= corr;

      // ---- repack P (C-layout) into a B-fragment via xor-16 shuffles ----
      union H2U { _Float16 h[2]; unsigned int u; };
      unsigned int own1[4], own2[4];
#pragma unroll
      for (int i = 0; i < 4; ++i) {
        H2U a; a.h[0] = ph[2 * i];     a.h[1] = ph[2 * i + 1]; own1[i] = a.u;
        H2U b; b.h[0] = ph[8 + 2 * i]; b.h[1] = ph[9 + 2 * i]; own2[i] = b.u;
      }
      union VP { v16h v; unsigned int u[8]; } pb;
#pragma unroll
      for (int i = 0; i < 4; ++i) {
        unsigned int o1 = (unsigned int)__shfl_xor((int)own1[i], 16);
        unsigned int o2 = (unsigned int)__shfl_xor((int)own2[i], 16);
        pb.u[i]     = hh ? o2      : own1[i];  // keys 0..7  / 16..23
        pb.u[4 + i] = hh ? own2[i] : o1;       // keys 8..15 / 24..31
      }

      // ---- O^T += V^T * P : 4 dh tiles, 1 wmma each ----
#pragma unroll
      for (int d = 0; d < 4; ++d) {
        v16h vf = frag_a16(&Vts[(d * 16 + l15) * 40], hh * 8);
        accO[d] = __builtin_amdgcn_wmma_f32_16x16x32_f16(
            false, vf, false, pb.v, (short)0, accO[d], false, false);
      }
    }
    __syncthreads();   // protect Ks/Vts before next stage
  }

  // ---- normalize + store O as (B,T,H,DH) f16 (== row-major (B*T, D)) ----
  float inv = 1.f / (lrun + 1e-8f);
  const int b = bh >> 4, h = bh & (NHEAD - 1);
#pragma unroll
  for (int d = 0; d < 4; ++d)
#pragma unroll
    for (int r = 0; r < 8; ++r) {
      int dh = d * 16 + hh * 8 + r;
      Og[(((size_t)b * TSEQ + qcol) * NHEAD + h) * DHEAD + dh] = (_Float16)(accO[d][r] * inv);
    }
}

// ---------------------------------------------------------------------------
// launch
// ---------------------------------------------------------------------------
extern "C" void kernel_launch(void* const* d_in, const int* in_sizes, int n_in,
                              void* d_out, int out_size, void* d_ws, size_t ws_size,
                              hipStream_t stream)
{
  const float* x      = (const float*)d_in[0];
  const float* Wq     = (const float*)d_in[1];
  const float* Wk     = (const float*)d_in[2];
  const float* Wv     = (const float*)d_in[3];
  const float* Wo_w   = (const float*)d_in[4];
  const float* Wo_b   = (const float*)d_in[5];
  const float* span_w = (const float*)d_in[6];
  const float* span_b = (const float*)d_in[7];
  (void)in_sizes; (void)n_in; (void)out_size; (void)ws_size;

  char* ws = (char*)d_ws;
  size_t off = 0;
  auto walloc = [&](size_t bytes) -> void* {
    void* p = ws + off;
    off += (bytes + 255) & ~(size_t)255;
    return p;
  };
  const size_t NX = (size_t)MTOT * D_MODEL;     // 4M
  const size_t NW = (size_t)D_MODEL * D_MODEL;  // 1M
  _Float16* x16  = (_Float16*)walloc(NX * 2);
  _Float16* Wq16 = (_Float16*)walloc(NW * 2);
  _Float16* Wk16 = (_Float16*)walloc(NW * 2);
  _Float16* Wv16 = (_Float16*)walloc(NW * 2);
  _Float16* Wo16 = (_Float16*)walloc(NW * 2);
  _Float16* q16  = (_Float16*)walloc(NX * 2);   // (B,H,T,DH)
  _Float16* k16  = (_Float16*)walloc(NX * 2);   // (B,H,T,DH)
  _Float16* vt16 = (_Float16*)walloc(NX * 2);   // (B,H,DH,T)
  _Float16* o16  = (_Float16*)walloc(NX * 2);   // (B,T,H,DH)
  float* xmean   = (float*)walloc((size_t)BATCH * D_MODEL * 4);
  float* zbuf    = (float*)walloc((size_t)BATCH * NHEAD * 4);

  f32_to_f16<<<(int)((NX + 255) / 256), 256, 0, stream>>>(x, x16, (int)NX);
  f32_to_f16<<<(int)((NW + 255) / 256), 256, 0, stream>>>(Wq, Wq16, (int)NW);
  f32_to_f16<<<(int)((NW + 255) / 256), 256, 0, stream>>>(Wk, Wk16, (int)NW);
  f32_to_f16<<<(int)((NW + 255) / 256), 256, 0, stream>>>(Wv, Wv16, (int)NW);
  f32_to_f16<<<(int)((NW + 255) / 256), 256, 0, stream>>>(Wo_w, Wo16, (int)NW);

  colmean_kernel<<<(BATCH * D_MODEL + 255) / 256, 256, 0, stream>>>(x, xmean);
  span_z_kernel<<<1, 32, 0, stream>>>(xmean, span_w, span_b, zbuf);

  dim3 gg(MTOT / 128, D_MODEL / 128);
  gemm128_wmma<1><<<gg, 256, 0, stream>>>(x16, Wq16, nullptr, q16);
  gemm128_wmma<1><<<gg, 256, 0, stream>>>(x16, Wk16, nullptr, k16);
  gemm128_wmma<2><<<gg, 256, 0, stream>>>(x16, Wv16, nullptr, vt16);

  attn_span_wmma<<<dim3(TSEQ / 128, BATCH * NHEAD), 256, 0, stream>>>(q16, k16, vt16, zbuf, o16);

  gemm128_wmma<3><<<gg, 256, 0, stream>>>(o16, Wo16, Wo_b, (float*)d_out);
}
